// QNetwork_19241453486367
// MI455X (gfx1250) — matrix operand order repaired and checked by reading, not compiled, not run on hardware
//
#include <hip/hip_runtime.h>
#include <hip/hip_bf16.h>

typedef __attribute__((ext_vector_type(16))) _Float16 v16h;
typedef __attribute__((ext_vector_type(8)))  _Float16 h8;
typedef __attribute__((ext_vector_type(8)))  float    v8f;

#define LRELU(z) ((z) >= 0.f ? (z) : 0.01f * (z))

// ---------------------------------------------------------------------------
// WMMA fragment helpers (wave32, v_wmma_f32_16x16x32_f16)
//
// A (16x32 f16): lane (l16, half): M = l16.
//   elements 0..7  -> K = k0 + 8*half + e        (16B load)
//   elements 8..15 -> K = k0 + 16 + 8*half + e-8 (16B load)
// B (32x16 f16): lane (l16, half): N = l16.
//   elements 0..7  -> K = k0 + 8*half + e        (16B load)
//   elements 8..15 -> K = k0 + 16 + 8*half + e-8 (16B load)
// D (16x16 f32): element r -> M = r + 8*half, N = l16.
// ---------------------------------------------------------------------------
__device__ __forceinline__ v16h frag_a(const _Float16* p, int ld, int row,
                                       int k0, int half) {
  const _Float16* q = p + row * ld + k0 + 8 * half;
  h8 lo = *(const h8*)(q);
  h8 hi = *(const h8*)(q + 16);
  return __builtin_shufflevector(lo, hi, 0, 1, 2, 3, 4, 5, 6, 7,
                                 8, 9, 10, 11, 12, 13, 14, 15);
}

__device__ __forceinline__ v16h frag_b(const _Float16* p, int ld, int col,
                                       int k0, int half) {
  const _Float16* q = p + col * ld + k0 + 8 * half;
  h8 lo = *(const h8*)(q);
  h8 hi = *(const h8*)(q + 16);
  return __builtin_shufflevector(lo, hi, 0, 1, 2, 3, 4, 5, 6, 7,
                                 8, 9, 10, 11, 12, 13, 14, 15);
}

__device__ __forceinline__ v8f wmma_f16(v16h a, v16h b, v8f c) {
  return __builtin_amdgcn_wmma_f32_16x16x32_f16(false, a, false, b,
                                                (short)0, c, false, false);
}

// ---------------------------------------------------------------------------
// Parallel in-place LayerNorm + LeakyReLU over 64 rows x 256 f16 cols in LDS.
// All 128 threads active: 2 threads per row (t, t^1 in the same wave32),
// 16-byte vector LDS accesses, pair reduction via shfl_xor.
// ---------------------------------------------------------------------------
__device__ __forceinline__ void ln_lrelu_64x256(_Float16* hBuf, int ldh,
                                                const float* __restrict__ g,
                                                const float* __restrict__ bi,
                                                int t) {
  const int row = t >> 1;
  const int hcol = t & 1;          // which 128-col half this thread owns
  _Float16* hr = hBuf + row * ldh + hcol * 128;
  float s = 0.f, s2 = 0.f;
#pragma unroll
  for (int i = 0; i < 16; ++i) {
    h8 v = *(const h8*)(hr + i * 8);
#pragma unroll
    for (int j = 0; j < 8; ++j) {
      float y = (float)v[j];
      s += y;
      s2 += y * y;
    }
  }
  s  += __shfl_xor(s, 1, 32);
  s2 += __shfl_xor(s2, 1, 32);
  float m = s * (1.f / 256.f);
  float var = s2 * (1.f / 256.f) - m * m;
  float inv = rsqrtf(var + 1e-5f);
  const float* gg = g + hcol * 128;
  const float* bb = bi + hcol * 128;
#pragma unroll
  for (int i = 0; i < 16; ++i) {
    h8 v = *(const h8*)(hr + i * 8);
    h8 o;
#pragma unroll
    for (int j = 0; j < 8; ++j) {
      float y = (float)v[j];
      float z = (y - m) * inv * gg[i * 8 + j] + bb[i * 8 + j];
      o[j] = (_Float16)LRELU(z);
    }
    *(h8*)(hr + i * 8) = o;
  }
}

// ---------------------------------------------------------------------------
// Kernel 0: one-time per-launch weight conversion f32 -> f16 in workspace.
// fc2_w: 256x256, fc3_w: 128x256, f1_w: 512x264 zero-padded to 512x288.
// ---------------------------------------------------------------------------
__global__ __launch_bounds__(256)
void prep_weights(const float* __restrict__ w2, const float* __restrict__ w3,
                  const float* __restrict__ f1w,
                  _Float16* __restrict__ w2h, _Float16* __restrict__ w3h,
                  _Float16* __restrict__ f1h) {
  int i = blockIdx.x * 256 + threadIdx.x;
  const int N2 = 256 * 256;          // 65536
  const int N3 = 128 * 256;          // 32768
  const int NF = 512 * 288;          // 147456 (padded)
  if (i < N2) {
    w2h[i] = (_Float16)w2[i];
  } else if (i < N2 + N3) {
    int j = i - N2;
    w3h[j] = (_Float16)w3[j];
  } else if (i < N2 + N3 + NF) {
    int j = i - N2 - N3;
    int r = j / 288, c = j % 288;
    f1h[j] = (c < 264) ? (_Float16)f1w[r * 264 + c] : (_Float16)0.f;
  }
}

// ---------------------------------------------------------------------------
// Kernel 1: per batch element b (64 neighbour rows):
//   fc1 -> LN1 -> lrelu -> fc2 -> LN2 -> lrelu -> fc3 -> mask/pool/gather
//   -> emit padded f16 fc_in row (288) to workspace.
// Block = 128 threads = 4 wave32s. Grid = B = 4096.
// ---------------------------------------------------------------------------
__global__ __launch_bounds__(128)
void qnet_stage1(const float* __restrict__ x_feats, const int* __restrict__ actions,
                 const float* __restrict__ fc1_w, const float* __restrict__ fc1_b,
                 const float* __restrict__ ln1_g, const float* __restrict__ ln1_b,
                 const _Float16* __restrict__ w2h, const float* __restrict__ fc2_b,
                 const float* __restrict__ ln2_g, const float* __restrict__ ln2_b,
                 const _Float16* __restrict__ w3h, const float* __restrict__ fc3_b,
                 _Float16* __restrict__ fcinh) {
  constexpr int LDH = 264;                    // 256 + 8 f16 pad (bank spread)
  __shared__ _Float16 hA[64 * LDH];           // 33.8 KB, reused h1 -> h2
  __shared__ float s_mask[64];
  __shared__ float s_colsum[128];
  __shared__ float s_chosen[128];
  __shared__ float s_cnt;

  const int b = blockIdx.x;
  const int t = threadIdx.x;
  const int wv = t >> 5;
  const int lane = t & 31;
  const int half = lane >> 4;
  const int l16 = lane & 15;
  const float* xrow = x_feats + (size_t)b * 323;
  const int act = actions[b];

  // ---- neighbour mask -----------------------------------------------------
  if (t < 64) {
    const float* nb = xrow + 3 + t * 5;
    float m = 0.f;
#pragma unroll
    for (int j = 0; j < 5; ++j)
      if (nb[j] != 0.f) m = 1.f;
    s_mask[t] = m;
  }
  __syncthreads();

  // ---- fc1: (64x8) @ fc1_w^T (8x256), K zero-padded to 32 -----------------
  {
#pragma unroll
    for (int mt = 0; mt < 4; ++mt) {
      v16h a = {};
      if (half == 0) {                         // only K=0..7 nonzero
        int row = mt * 16 + l16;
        const float* nb = xrow + 3 + row * 5;
        a[0] = (_Float16)xrow[0];
        a[1] = (_Float16)xrow[1];
        a[2] = (_Float16)xrow[2];
#pragma unroll
        for (int j = 0; j < 5; ++j) a[3 + j] = (_Float16)nb[j];
      }
#pragma unroll
      for (int ntl = 0; ntl < 4; ++ntl) {
        int col0 = (wv * 4 + ntl) * 16;
        v16h bb = {};
        if (half == 0) {                       // B: K = 16*half + e
          int n = col0 + l16;
#pragma unroll
          for (int e = 0; e < 8; ++e) bb[e] = (_Float16)fc1_w[n * 8 + e];
        }
        v8f d = {};
        d = wmma_f16(a, bb, d);
        int cc = col0 + l16;
        float bias = fc1_b[cc];
#pragma unroll
        for (int r = 0; r < 8; ++r)
          hA[(mt * 16 + r + 8 * half) * LDH + cc] = (_Float16)(d[r] + bias);
      }
    }
  }
  __syncthreads();

  // ---- LN1 + lrelu in place (all 128 threads) -----------------------------
  ln_lrelu_64x256(hA, LDH, ln1_g, ln1_b, t);
  if (t == 0) {            // tiny serial count, hidden behind barrier wait
    float c = 0.f;
    for (int i = 0; i < 64; ++i) c += s_mask[i];
    s_cnt = fmaxf(c, 1.f);
  }
  __syncthreads();

  // ---- fc2: (64x256) @ w2^T (256x256); accumulate in regs, rewrite hA -----
  {
    v8f acc[16] = {};
#pragma unroll
    for (int mt = 0; mt < 4; ++mt) {
      v16h afr[8];
#pragma unroll
      for (int kc = 0; kc < 8; ++kc)
        afr[kc] = frag_a(hA, LDH, mt * 16 + l16, kc * 32, half);
#pragma unroll
      for (int ntl = 0; ntl < 4; ++ntl) {
        int n = (wv * 4 + ntl) * 16 + l16;
#pragma unroll
        for (int kc = 0; kc < 8; ++kc) {
          v16h bb = frag_b(w2h, 256, n, kc * 32, half);
          acc[mt * 4 + ntl] = wmma_f16(afr[kc], bb, acc[mt * 4 + ntl]);
        }
      }
    }
    __syncthreads();  // everyone done reading hA (h1)
#pragma unroll
    for (int mt = 0; mt < 4; ++mt) {
#pragma unroll
      for (int ntl = 0; ntl < 4; ++ntl) {
        int cc = (wv * 4 + ntl) * 16 + l16;
        float bias = fc2_b[cc];
#pragma unroll
        for (int r = 0; r < 8; ++r)
          hA[(mt * 16 + r + 8 * half) * LDH + cc] =
              (_Float16)(acc[mt * 4 + ntl][r] + bias);
      }
    }
  }
  __syncthreads();

  // ---- LN2 + lrelu in place ----------------------------------------------
  ln_lrelu_64x256(hA, LDH, ln2_g, ln2_b, t);
  __syncthreads();

  // ---- fc3: (64x256) @ w3^T (256x128) + mask-pool + chosen gather ---------
  {
    v8f e3[8] = {};
#pragma unroll
    for (int mt = 0; mt < 4; ++mt) {
      v16h afr[8];
#pragma unroll
      for (int kc = 0; kc < 8; ++kc)
        afr[kc] = frag_a(hA, LDH, mt * 16 + l16, kc * 32, half);
#pragma unroll
      for (int ntl = 0; ntl < 2; ++ntl) {
        int n = (wv * 2 + ntl) * 16 + l16;
#pragma unroll
        for (int kc = 0; kc < 8; ++kc) {
          v16h bb = frag_b(w3h, 256, n, kc * 32, half);
          e3[mt * 2 + ntl] = wmma_f16(afr[kc], bb, e3[mt * 2 + ntl]);
        }
      }
    }
#pragma unroll
    for (int ntl = 0; ntl < 2; ++ntl) {
      int col = (wv * 2 + ntl) * 16 + l16;   // unique col per wave: no races
      float bias = fc3_b[col];
      float part = 0.f;
#pragma unroll
      for (int mt = 0; mt < 4; ++mt) {
#pragma unroll
        for (int r = 0; r < 8; ++r) {
          int rl = mt * 16 + r + 8 * half;
          float val = e3[mt * 2 + ntl][r] + bias;
          part += s_mask[rl] * val;
          if (rl == act) s_chosen[col] = val;  // exactly one writer
        }
      }
      part += __shfl_xor(part, 16, 32);        // lanes L, L+16 share col
      if (half == 0) s_colsum[col] = part;     // deterministic, no atomics
    }
  }
  __syncthreads();

  // ---- emit fc_in row: [pos(3) | chosen(128) | pooled(128) | state(5) | 0]
  {
    _Float16* orow = fcinh + (size_t)b * 288;
    float invc = 1.f / s_cnt;
    for (int c = t; c < 288; c += 128) {
      float v;
      if (c < 3)        v = xrow[c];
      else if (c < 131) v = s_chosen[c - 3];
      else if (c < 259) v = s_colsum[c - 131] * invc;
      else if (c < 264) v = xrow[3 + act * 5 + (c - 259)];
      else              v = 0.f;
      orow[c] = (_Float16)v;
    }
  }
}

// ---------------------------------------------------------------------------
// Kernel 2: f1 (264->512 as padded K=288 WMMA GEMM) + LNf + lrelu + f2 dot.
// Block = 128 threads (4 waves) handles 16 batch rows. Grid = 256.
// ---------------------------------------------------------------------------
__global__ __launch_bounds__(128)
void qnet_stage2(const _Float16* __restrict__ fcinh, const _Float16* __restrict__ f1h,
                 const float* __restrict__ f1_b, const float* __restrict__ lnf_g,
                 const float* __restrict__ lnf_b, const float* __restrict__ f2_w,
                 const float* __restrict__ f2_b, float* __restrict__ out) {
  __shared__ float ytile[16 * 512];  // 32 KB

  const int t = threadIdx.x;
  const int wv = t >> 5;
  const int lane = t & 31;
  const int half = lane >> 4;
  const int l16 = lane & 15;
  const int b0 = blockIdx.x * 16;

  const _Float16* A = fcinh + (size_t)b0 * 288;

  v16h afr[9];
#pragma unroll
  for (int kc = 0; kc < 9; ++kc)
    afr[kc] = frag_a(A, 288, l16, kc * 32, half);

#pragma unroll
  for (int ntl = 0; ntl < 8; ++ntl) {
    int col0 = (wv * 8 + ntl) * 16;
    v8f d = {};
#pragma unroll
    for (int kc = 0; kc < 9; ++kc) {
      v16h bb = frag_b(f1h, 288, col0 + l16, kc * 32, half);
      d = wmma_f16(afr[kc], bb, d);
    }
    int col = col0 + l16;
    float bias = f1_b[col];
#pragma unroll
    for (int r = 0; r < 8; ++r)
      ytile[(r + 8 * half) * 512 + col] = d[r] + bias;
  }
  __syncthreads();

  // ---- LNf + lrelu + f2 dot: 8 threads per row, float4 loads --------------
  {
    const int row = t >> 3;
    const int sub = t & 7;                 // 64-col chunk owner
    const float* y = ytile + row * 512 + sub * 64;
    float s = 0.f, s2 = 0.f;
#pragma unroll
    for (int i = 0; i < 16; ++i) {
      float4 v = *(const float4*)(y + i * 4);
      s  += v.x + v.y + v.z + v.w;
      s2 += v.x * v.x + v.y * v.y + v.z * v.z + v.w * v.w;
    }
    s  += __shfl_xor(s, 1, 32);
    s  += __shfl_xor(s, 2, 32);
    s  += __shfl_xor(s, 4, 32);
    s2 += __shfl_xor(s2, 1, 32);
    s2 += __shfl_xor(s2, 2, 32);
    s2 += __shfl_xor(s2, 4, 32);
    float m = s * (1.f / 512.f);
    float var = s2 * (1.f / 512.f) - m * m;
    float inv = rsqrtf(var + 1e-5f);
    const float* g  = lnf_g + sub * 64;
    const float* bb = lnf_b + sub * 64;
    const float* w  = f2_w + sub * 64;
    float acc = 0.f;
#pragma unroll
    for (int i = 0; i < 64; ++i) {
      float z = (y[i] - m) * inv * g[i] + bb[i];
      z = LRELU(z);
      acc += z * w[i];
    }
    acc += __shfl_xor(acc, 1, 32);
    acc += __shfl_xor(acc, 2, 32);
    acc += __shfl_xor(acc, 4, 32);
    if (sub == 0) out[b0 + row] = acc + f2_b[0];
  }
}

// ---------------------------------------------------------------------------
extern "C" void kernel_launch(void* const* d_in, const int* in_sizes, int n_in,
                              void* d_out, int out_size, void* d_ws, size_t ws_size,
                              hipStream_t stream) {
  (void)in_sizes; (void)n_in; (void)out_size; (void)ws_size;

  const float* x_feats = (const float*)d_in[0];
  const int*   actions = (const int*)d_in[1];
  const float* fc1_w = (const float*)d_in[2];
  const float* fc1_b = (const float*)d_in[3];
  const float* ln1_g = (const float*)d_in[4];
  const float* ln1_b = (const float*)d_in[5];
  const float* fc2_w = (const float*)d_in[6];
  const float* fc2_b = (const float*)d_in[7];
  const float* ln2_g = (const float*)d_in[8];
  const float* ln2_b = (const float*)d_in[9];
  const float* fc3_w = (const float*)d_in[10];
  const float* fc3_b = (const float*)d_in[11];
  const float* f1_w  = (const float*)d_in[12];
  const float* f1_b  = (const float*)d_in[13];
  const float* lnf_g = (const float*)d_in[14];
  const float* lnf_b = (const float*)d_in[15];
  const float* f2_w  = (const float*)d_in[16];
  const float* f2_b  = (const float*)d_in[17];
  float* out = (float*)d_out;

  // Workspace layout (all 16B+ aligned from a 256B-aligned base):
  _Float16* w2h   = (_Float16*)d_ws;        // 256*256 f16 = 131072 B
  _Float16* w3h   = w2h + 256 * 256;        // 128*256 f16 =  65536 B
  _Float16* f1h   = w3h + 128 * 256;        // 512*288 f16 = 294912 B
  _Float16* fcinh = f1h + 512 * 288;        // 4096*288 f16 = 2359296 B

  // 65536 + 32768 + 147456 = 245760 elements = 960 * 256
  prep_weights<<<960, 256, 0, stream>>>(fc2_w, fc3_w, f1_w, w2h, w3h, f1h);
  qnet_stage1<<<4096, 128, 0, stream>>>(x_feats, actions, fc1_w, fc1_b, ln1_g,
                                        ln1_b, w2h, fc2_b, ln2_g, ln2_b, w3h,
                                        fc3_b, fcinh);
  qnet_stage2<<<256, 128, 0, stream>>>(fcinh, f1h, f1_b, lnf_g, lnf_b, f2_w,
                                       f2_b, out);
}
